// Res_Memory_66529043415390
// MI455X (gfx1250) — matrix operand order repaired and checked by reading, not compiled, not run on hardware
//
#include <hip/hip_runtime.h>
#include <hip/hip_bf16.h>

// ---------------------------------------------------------------------------
// Problem constants (from reference): B=16, IN=512, HID=512, M=1024, N=4096
// ---------------------------------------------------------------------------
#define BATCH   16
#define IN_DIM  512
#define HID     512
#define MSLOTS  1024
#define NPTS    4096

typedef __bf16 bf16_t;
typedef __attribute__((ext_vector_type(16))) __bf16        v16bf;
typedef __attribute__((ext_vector_type(8)))  float         v8f;
typedef __attribute__((ext_vector_type(4)))  unsigned int  u32x4;

// B-panel in LDS: [BN columns][K elements], K-contiguous, padded for banks
#define BN   32
#define LDK  520   // 512 + 8 bf16 pad -> row stride 1040 B (16B aligned)

union Frag16 { v16bf v; u32x4 q[2]; };
union Pack8  { u32x4 q; bf16_t h[8]; };

// ---------------------------------------------------------------------------
// CDNA5 async global->LDS copy (ASYNCcnt path), 16 bytes per issue.
// generic shared-pointer low 32 bits give the
// in-allocation LDS byte offset on AMDGPU.
// ---------------------------------------------------------------------------
__device__ __forceinline__ void async_copy_b128(unsigned lds_off, const void* gptr) {
    asm volatile("global_load_async_to_lds_b128 %0, %1, off"
                 :: "v"(lds_off), "v"((unsigned long long)gptr)
                 : "memory");
}
__device__ __forceinline__ void wait_async0() {
    asm volatile("s_wait_asynccnt 0x0" ::: "memory");
}

// A fragment (16x32 bf16, row-major source, stride lda):
// lane l<16: M=l,  K = {k0..k0+7, k0+16..k0+23}
// lane l>=16: M=l-16, K = {k0+8..k0+15, k0+24..k0+31}
__device__ __forceinline__ v16bf load_a_frag(const bf16_t* __restrict__ A, int lda,
                                             int m0, int k0, int lane) {
    Frag16 f;
    const int m     = m0 + (lane & 15);
    const int khalf = (lane >> 4) << 3;
    const bf16_t* p = A + (size_t)m * lda + k0 + khalf;
    f.q[0] = *(const u32x4*)(p);
    f.q[1] = *(const u32x4*)(p + 16);
    return f.v;
}

// B fragment (32x16 bf16) from LDS panel stored as [n][k] (k contiguous):
// lane l: N = l&15, K = k0 + 8*(l>>4) + {0..7, 16..23}
__device__ __forceinline__ v16bf load_b_frag_lds(const bf16_t* __restrict__ L,
                                                 int nloc, int k0, int lane) {
    Frag16 f;
    const int n  = nloc + (lane & 15);
    const int kb = k0 + ((lane >> 4) << 3);
    const bf16_t* p = L + (size_t)n * LDK + kb;
    f.q[0] = *(const u32x4*)(p);
    f.q[1] = *(const u32x4*)(p + 16);
    return f.v;
}

// ---------------------------------------------------------------------------
// Prep 1: row softmax of m1 -> m_soft (bf16) and ent[m] = sum q*log q
//         ent = (sum e*(x-mx))/S - log S
// ---------------------------------------------------------------------------
__global__ __launch_bounds__(256) void softmax_prep_kernel(
        const float* __restrict__ m1, bf16_t* __restrict__ msoftb,
        float* __restrict__ ent) {
    __shared__ float red[256];
    const int m = blockIdx.x;
    const int t = threadIdx.x;
    const float* row = m1 + (size_t)m * HID;
    const float x0 = row[t], x1 = row[t + 256];

    float mx = fmaxf(x0, x1);
    red[t] = mx; __syncthreads();
    for (int s = 128; s > 0; s >>= 1) { if (t < s) red[t] = fmaxf(red[t], red[t + s]); __syncthreads(); }
    mx = red[0]; __syncthreads();

    const float e0 = __expf(x0 - mx), e1 = __expf(x1 - mx);
    red[t] = e0 + e1; __syncthreads();
    for (int s = 128; s > 0; s >>= 1) { if (t < s) red[t] += red[t + s]; __syncthreads(); }
    const float S = red[0]; __syncthreads();

    red[t] = e0 * (x0 - mx) + e1 * (x1 - mx); __syncthreads();
    for (int s = 128; s > 0; s >>= 1) { if (t < s) red[t] += red[t + s]; __syncthreads(); }
    const float W = red[0];

    const float inv = 1.0f / S;
    msoftb[(size_t)m * HID + t]       = (bf16_t)(e0 * inv);
    msoftb[(size_t)m * HID + t + 256] = (bf16_t)(e1 * inv);
    if (t == 0) ent[m] = W * inv - __logf(S);
}

// ---------------------------------------------------------------------------
// Prep 2: convert w1 [HID,IN] and w2 [HID,2*HID] to bf16
// ---------------------------------------------------------------------------
__global__ __launch_bounds__(256) void cvt_weights_kernel(
        const float* __restrict__ w1, const float* __restrict__ w2,
        bf16_t* __restrict__ w1b, bf16_t* __restrict__ w2b) {
    const int i = blockIdx.x * 256 + threadIdx.x;
    if (i < HID * IN_DIM)  w1b[i] = (bf16_t)w1[i];
    if (i < HID * 2 * HID) w2b[i] = (bf16_t)w2[i];
}

// ---------------------------------------------------------------------------
// Kernel 1: emb = relu(w1 @ x[b]) , stored bf16 [B][N][HID]  (h contiguous!)
// Block: 128 h-rows x 32 n-cols, 8 waves (one 16-row strip each), full K in LDS
// ---------------------------------------------------------------------------
__global__ __launch_bounds__(256) void conv1_kernel(
        const float* __restrict__ x, const bf16_t* __restrict__ w1b,
        const float* __restrict__ b1, bf16_t* __restrict__ embb) {
    __shared__ bf16_t ldsB[BN * LDK];
    const int n0 = blockIdx.x * BN;
    const int m0 = blockIdx.y * 128;
    const int bz = blockIdx.z;
    const int t = threadIdx.x, lane = t & 31, wv = t >> 5;

    const float* xb = x + (size_t)bz * IN_DIM * NPTS;
    {   // stage x[k][n0..] -> ldsB[n][k] as bf16 (coalesced global reads)
        const int n = t & 31;
        for (int r = (t >> 5); r < IN_DIM; r += 8)
            ldsB[n * LDK + r] = (bf16_t)xb[(size_t)r * NPTS + n0 + n];
    }
    __syncthreads();

    v8f acc[2] = {};
    const int mrow = m0 + wv * 16;
#pragma unroll
    for (int ks = 0; ks < IN_DIM / 32; ++ks) {
        const int k0 = ks * 32;
        v16bf a = load_a_frag(w1b, IN_DIM, mrow, k0, lane);
#pragma unroll
        for (int j = 0; j < 2; ++j) {
            v16bf bb = load_b_frag_lds(ldsB, j * 16, k0, lane);
            acc[j] = __builtin_amdgcn_wmma_f32_16x16x32_bf16(
                         false, a, false, bb, (short)0, acc[j], false, false);
        }
    }
    // pack 8 consecutive h per lane -> one 16B store per fragment
    const int mh = mrow + ((lane >> 4) << 3);
    const int nc = n0 + (lane & 15);
#pragma unroll
    for (int j = 0; j < 2; ++j) {
        Pack8 pk;
#pragma unroll
        for (int e = 0; e < 8; ++e) {
            float v = acc[j][e] + b1[mh + e];
            v = v > 0.0f ? v : 0.0f;
            pk.h[e] = (bf16_t)v;
        }
        *(u32x4*)(embb + ((size_t)bz * NPTS + nc + j * 16) * HID + mh) = pk.q;
    }
}

// ---------------------------------------------------------------------------
// Kernel 2: idx[b,n] = argmin_m ( ent[m] - m_soft @ emb[:,n] )
// Block: 32 n-cols, loops all M=1024 in 128-row chunks (one 16-row strip/wave)
// Panel staged with CDNA5 async global->LDS copies (contiguous bf16 rows).
// ---------------------------------------------------------------------------
__global__ __launch_bounds__(256) void score_argmin_kernel(
        const bf16_t* __restrict__ embb, const bf16_t* __restrict__ msoftb,
        const float* __restrict__ ent, int* __restrict__ idx_out) {
    __shared__ bf16_t ldsB[BN * LDK];
    __shared__ float redv[8][BN];
    __shared__ int   redi[8][BN];
    const int n0 = blockIdx.x * BN;
    const int bz = blockIdx.y;
    const int t = threadIdx.x, lane = t & 31, wv = t >> 5;

    const bf16_t* eb = embb + (size_t)bz * NPTS * HID;
    {   // async stage emb[n0+n][0..511] -> ldsB[n][*] : 8 threads per row
        const int n = t >> 3;
        const unsigned lbase = (unsigned)(size_t)&ldsB[n * LDK];
        const char* grow = (const char*)(eb + (size_t)(n0 + n) * HID);
#pragma unroll
        for (int i = 0; i < 8; ++i) {
            const int off = ((t & 7) + i * 8) * 16;   // 64 x 16B per 1KB row
            async_copy_b128(lbase + off, grow + off);
        }
        wait_async0();
    }
    __syncthreads();

    float minv[2]; int mini[2];
#pragma unroll
    for (int j = 0; j < 2; ++j) { minv[j] = 3.4e38f; mini[j] = 0; }
    const int lhalf = (lane >> 4) << 3;

    for (int mc = 0; mc < MSLOTS / 128; ++mc) {
        const int mbase = mc * 128 + wv * 16;
        v8f acc[2] = {};
#pragma unroll
        for (int ks = 0; ks < HID / 32; ++ks) {
            const int k0 = ks * 32;
            v16bf a = load_a_frag(msoftb, HID, mbase, k0, lane);
#pragma unroll
            for (int j = 0; j < 2; ++j) {
                v16bf bb = load_b_frag_lds(ldsB, j * 16, k0, lane);
                acc[j] = __builtin_amdgcn_wmma_f32_16x16x32_bf16(
                             false, a, false, bb, (short)0, acc[j], false, false);
            }
        }
        float ent8[8];
#pragma unroll
        for (int e = 0; e < 8; ++e) ent8[e] = ent[mbase + lhalf + e];
#pragma unroll
        for (int j = 0; j < 2; ++j)
#pragma unroll
            for (int e = 0; e < 8; ++e) {
                const float cand = ent8[e] - acc[j][e];
                const int   mm   = mbase + lhalf + e;
                if (cand < minv[j] || (cand == minv[j] && mm < mini[j])) {
                    minv[j] = cand; mini[j] = mm;
                }
            }
    }
    // combine lane pairs (l, l^16): same n column, other 8 m-rows
#pragma unroll
    for (int j = 0; j < 2; ++j) {
        const float ov = __shfl_xor(minv[j], 16, 32);
        const int   oi = __shfl_xor(mini[j], 16, 32);
        if (ov < minv[j] || (ov == minv[j] && oi < mini[j])) { minv[j] = ov; mini[j] = oi; }
    }
    if (lane < 16) {
#pragma unroll
        for (int j = 0; j < 2; ++j) {
            redv[wv][j * 16 + lane] = minv[j];
            redi[wv][j * 16 + lane] = mini[j];
        }
    }
    __syncthreads();
    if (t < BN) {
        float bv = redv[0][t]; int bi = redi[0][t];
#pragma unroll
        for (int w = 1; w < 8; ++w) {
            const float v = redv[w][t]; const int i = redi[w][t];
            if (v < bv || (v == bv && i < bi)) { bv = v; bi = i; }
        }
        idx_out[(size_t)bz * NPTS + n0 + t] = bi;
    }
}

// ---------------------------------------------------------------------------
// Kernel 3: out = w2 @ [emb ; emb - m1[idx]] + b2  (f32 output)
// Two K-phases reuse one LDS panel: phase0 = emb (async copy),
// phase1 = residual (gather + convert).
// ---------------------------------------------------------------------------
__global__ __launch_bounds__(256) void conv2_kernel(
        const bf16_t* __restrict__ embb, const float* __restrict__ m1,
        const int* __restrict__ idx_in, const bf16_t* __restrict__ w2b,
        const float* __restrict__ b2, float* __restrict__ out) {
    __shared__ bf16_t ldsB[BN * LDK];
    __shared__ int lds_idx[BN];
    const int n0 = blockIdx.x * BN;
    const int m0 = blockIdx.y * 128;
    const int bz = blockIdx.z;
    const int t = threadIdx.x, lane = t & 31, wv = t >> 5;

    const bf16_t* eb = embb + (size_t)bz * NPTS * HID;
    if (t < BN) lds_idx[t] = idx_in[(size_t)bz * NPTS + n0 + t];

    {   // phase 0 staging: async copy of emb panel (h contiguous rows)
        const int n = t >> 3;
        const unsigned lbase = (unsigned)(size_t)&ldsB[n * LDK];
        const char* grow = (const char*)(eb + (size_t)(n0 + n) * HID);
#pragma unroll
        for (int i = 0; i < 8; ++i) {
            const int off = ((t & 7) + i * 8) * 16;
            async_copy_b128(lbase + off, grow + off);
        }
        wait_async0();
    }
    __syncthreads();

    v8f acc[2] = {};
    const int mrow = m0 + wv * 16;
#pragma unroll
    for (int ks = 0; ks < HID / 32; ++ks) {
        const int k0 = ks * 32;
        v16bf a = load_a_frag(w2b, 2 * HID, mrow, k0, lane);
#pragma unroll
        for (int j = 0; j < 2; ++j) {
            v16bf bb = load_b_frag_lds(ldsB, j * 16, k0, lane);
            acc[j] = __builtin_amdgcn_wmma_f32_16x16x32_bf16(
                         false, a, false, bb, (short)0, acc[j], false, false);
        }
    }
    __syncthreads();   // everyone done reading phase-0 panel

    {   // phase 1 staging: residual r = emb - m1[idx[n]], 64 contiguous elems/thread
        const int n  = t >> 3;
        const int rb = (t & 7) * 64;
        const bf16_t* erow  = eb + (size_t)(n0 + n) * HID;
        const float*  mrowp = m1 + (size_t)lds_idx[n] * HID;
#pragma unroll 8
        for (int r = rb; r < rb + 64; ++r)
            ldsB[n * LDK + r] = (bf16_t)((float)erow[r] - mrowp[r]);
    }
    __syncthreads();

#pragma unroll
    for (int ks = 0; ks < HID / 32; ++ks) {
        const int k0 = ks * 32;
        v16bf a = load_a_frag(w2b, 2 * HID, mrow, HID + k0, lane);
#pragma unroll
        for (int j = 0; j < 2; ++j) {
            v16bf bb = load_b_frag_lds(ldsB, j * 16, k0, lane);
            acc[j] = __builtin_amdgcn_wmma_f32_16x16x32_bf16(
                         false, a, false, bb, (short)0, acc[j], false, false);
        }
    }

    const int mh = mrow + ((lane >> 4) << 3);
    const int nc = n0 + (lane & 15);
#pragma unroll
    for (int j = 0; j < 2; ++j)
#pragma unroll
        for (int e = 0; e < 8; ++e) {
            const int h = mh + e;
            out[((size_t)bz * HID + h) * NPTS + nc + j * 16] = acc[j][e] + b2[h];
        }
}

// ---------------------------------------------------------------------------
// Launch
// ---------------------------------------------------------------------------
extern "C" void kernel_launch(void* const* d_in, const int* in_sizes, int n_in,
                              void* d_out, int out_size, void* d_ws, size_t ws_size,
                              hipStream_t stream) {
    (void)in_sizes; (void)n_in; (void)out_size; (void)ws_size;

    const float* x  = (const float*)d_in[0];   // [B, IN, N, 1]
    const float* m1 = (const float*)d_in[1];   // [M, HID]
    const float* w1 = (const float*)d_in[2];   // [HID, IN]
    const float* b1 = (const float*)d_in[3];   // [HID]
    const float* w2 = (const float*)d_in[4];   // [HID, 2*HID]
    const float* b2 = (const float*)d_in[5];   // [HID]
    float* out = (float*)d_out;                // [B, HID, N, 1]

    // workspace layout (all offsets 1 KB aligned)
    char* ws = (char*)d_ws;
    size_t o = 0;
    bf16_t* w1b    = (bf16_t*)(ws + o); o += (size_t)HID * IN_DIM * 2;      //  512 KB
    bf16_t* w2b    = (bf16_t*)(ws + o); o += (size_t)HID * 2 * HID * 2;     // 1024 KB
    bf16_t* msoftb = (bf16_t*)(ws + o); o += (size_t)MSLOTS * HID * 2;      // 1024 KB
    float*  ent    = (float*)(ws + o);  o += (size_t)MSLOTS * 4;            //    4 KB
    int*    idx    = (int*)(ws + o);    o += (size_t)BATCH * NPTS * 4;      //  256 KB
    bf16_t* embb   = (bf16_t*)(ws + o); o += (size_t)BATCH * NPTS * HID * 2;//   64 MB

    softmax_prep_kernel<<<MSLOTS, 256, 0, stream>>>(m1, msoftb, ent);
    cvt_weights_kernel<<<(HID * 2 * HID + 255) / 256, 256, 0, stream>>>(w1, w2, w1b, w2b);
    conv1_kernel<<<dim3(NPTS / BN, HID / 128, BATCH), 256, 0, stream>>>(x, w1b, b1, embb);
    score_argmin_kernel<<<dim3(NPTS / BN, BATCH), 256, 0, stream>>>(embb, msoftb, ent, idx);
    conv2_kernel<<<dim3(NPTS / BN, HID / 128, BATCH), 256, 0, stream>>>(embb, m1, idx, w2b, b2, out);
}
